// VectorQuantizer2D_13907104105085
// MI455X (gfx1250) — matrix operand order repaired and checked by reading, not compiled, not run on hardware
//
#include <hip/hip_runtime.h>
#include <hip/hip_bf16.h>

typedef __attribute__((ext_vector_type(16))) _Float16 v16h;
typedef __attribute__((ext_vector_type(8)))  _Float16 v8h;
typedef __attribute__((ext_vector_type(8)))  float    v8f;

#define NCODE 4096
#define DIM   256
#define HWSZ  4096          // 64*64
#define NTOK  32768         // 8*64*64
#define ROWS  128           // tokens per block
#define KITER 64            // codes per K-tile iteration
#define NITER (NCODE / KITER)           // 64
#define CPAD  8             // cbt row = 72 halves = 144 B (16B-aligned rows)
#define ZPAD  8             // zt row  = 264 halves = 528 B (16B-aligned rows)
#define ZQ_ELEMS ((size_t)NTOK * DIM)   // 8,388,608 floats of z_q_st
// d_out layout (reference returns (z_q_st, vq_loss, idx)):
//   [0, ZQ_ELEMS)            z_q_st, [B,D,H,W] layout
//   [ZQ_ELEMS]               vq_loss scalar
//   [ZQ_ELEMS+1, +NTOK)      idx, flattened [B,H,W] (stored as float values)

// ---------------------------------------------------------------------------
// Prep: fp32 codebook -> TRANSPOSED f16 copy cbhT[dim][code] in workspace,
// per-code ||e||^2, zero loss slot. One block (64 threads) per code.
// The transposed layout makes each 64-code tile row 128 B contiguous, so the
// main kernel can stage it with global_load_async_to_lds_b128 (16 B / lane).
// ---------------------------------------------------------------------------
__global__ __launch_bounds__(64) void vq_prep(const float* __restrict__ cb,
                                              _Float16* __restrict__ cbhT,
                                              float* __restrict__ e2,
                                              float* __restrict__ loss_slot) {
    const int code = blockIdx.x;
    const int t = threadIdx.x;          // 0..63
    const float4* src = (const float4*)(cb + (size_t)code * DIM);
    float4 v = src[t];                  // dims 4t..4t+3
    _Float16* col = cbhT + code;        // column `code`, stride NCODE per dim
    col[(size_t)(4 * t + 0) * NCODE] = (_Float16)v.x;
    col[(size_t)(4 * t + 1) * NCODE] = (_Float16)v.y;
    col[(size_t)(4 * t + 2) * NCODE] = (_Float16)v.z;
    col[(size_t)(4 * t + 3) * NCODE] = (_Float16)v.w;
    float s = v.x * v.x + v.y * v.y + v.z * v.z + v.w * v.w;
    #pragma unroll
    for (int off = 16; off > 0; off >>= 1) s += __shfl_down(s, off);
    __shared__ float ws2[2];
    if ((t & 31) == 0) ws2[t >> 5] = s;
    __syncthreads();
    if (t == 0) e2[code] = ws2[0] + ws2[1];
    if (code == 0 && t == 0) *loss_slot = 0.0f;
}

// ---------------------------------------------------------------------------
// Async-stage one 64-code tile (32 KB) into an LDS buffer: 8 waves x 8 instrs,
// each lane copies 16 B (4 dim-rows x 8 chunks per wave-instruction).
// ---------------------------------------------------------------------------
__device__ __forceinline__ void stage_tile_async(size_t cbhTbytes, int it,
                                                 _Float16 (*buf)[KITER + CPAD],
                                                 int wave, int lane) {
    const int c8   = lane & 7;     // 16 B chunk within a 128 B dim-row
    const int dsub = lane >> 3;    // 0..3
    #pragma unroll
    for (int i = 0; i < 8; ++i) {
        const int d = wave * 32 + i * 4 + dsub;
        unsigned long long gaddr =
            (unsigned long long)(cbhTbytes + (size_t)d * (NCODE * 2)
                                 + (size_t)it * (KITER * 2) + c8 * 16);
        unsigned laddr = (unsigned)(size_t)(&buf[d][c8 * 8]);
        asm volatile("global_load_async_to_lds_b128 %0, %1, off"
                     :: "v"(laddr), "v"(gaddr) : "memory");
    }
}

// ---------------------------------------------------------------------------
// Main: fused GEMM (f16 WMMA, f32 accum) + argmin + gather + loss.
// 256 threads = 8 waves = 4 row-groups x 2 code-slots.
// Each wave: 32 rows x 32 codes via 4 accumulators (2 A x 2 B operands per
// k-step; A via ds_load_b128, B gathered by 16-bit element loads). Codebook tiles double-buffered via
// async-to-LDS so the copy overlaps the WMMA stream; one barrier/iteration.
// ---------------------------------------------------------------------------
__global__ __launch_bounds__(256) void vq_main(const float* __restrict__ z,
                                               const float* __restrict__ cb,
                                               const _Float16* __restrict__ cbhT,
                                               const float* __restrict__ e2g,
                                               float* __restrict__ out) {
    __shared__ __align__(16) _Float16 zt[ROWS][DIM + ZPAD];       // 67,584 B
    __shared__ __align__(16) _Float16 cbt[2][DIM][KITER + CPAD];  // 73,728 B
    __shared__ float redV[ROWS][2];
    __shared__ int   redI[ROWS][2];
    __shared__ int   fidx[ROWS];
    __shared__ float lred[8];

    const int tid    = threadIdx.x;
    const int lane   = tid & 31;
    const int wave   = tid >> 5;         // 0..7
    const int kslot  = wave & 1;         // code slot: 32 codes each
    const int mgroup = wave >> 1;        // row group: 32 rows each (0..3)
    const int p0     = blockIdx.x * ROWS;

    // ---- stage z tile (fp32 [B,D,H,W] -> f16 LDS [row][dim]) ----
    {
        const int r  = tid & 127;            // local row
        const int dg = tid >> 7;             // dim group 0..1 (128 dims each)
        const int p  = p0 + r;
        const int b  = p >> 12;              // p / 4096
        const int q  = p & (HWSZ - 1);
        const float* zrow = z + (size_t)b * DIM * HWSZ + q;
        #pragma unroll 4
        for (int i = 0; i < 128; ++i) {
            int d = dg * 128 + i;
            zt[r][d] = (_Float16)zrow[(size_t)d * HWSZ];
        }
    }

    float runMin[16];
    int   runIdx[16];
    #pragma unroll
    for (int j = 0; j < 16; ++j) { runMin[j] = 3.4e38f; runIdx[j] = 0x7fffffff; }

    const int arow0  = mgroup * 32 + (lane & 15);
    const int arow1  = arow0 + 16;
    const int khalf  = (lane >> 4) << 3;           // 0 or 8
    const int koff   = kslot * 32;
    const int cgLane = kslot * 32 + (lane & 15);   // per-lane code column
    const size_t cbhTbytes = (size_t)(const char*)cbhT;

    // prologue: async-stage tile 0 into buffer 0 (completion checked in-loop)
    stage_tile_async(cbhTbytes, 0, cbt[0], wave, lane);

    for (int it = 0; it < NITER; ++it) {
        asm volatile("s_wait_asynccnt 0x0" ::: "memory");  // my copies landed
        __syncthreads();            // everyone's copies landed; prev buf free
        if (it + 1 < NITER)         // overlap next tile copy with compute
            stage_tile_async(cbhTbytes, it + 1, cbt[(it + 1) & 1], wave, lane);
        if (it + 2 < NITER)         // warm L2 for the tile after that
            __builtin_prefetch((const char*)cbhT + (size_t)tid * (NCODE * 2)
                               + (size_t)(it + 2) * (KITER * 2), 0, 0);
        _Float16 (*cbc)[KITER + CPAD] = cbt[it & 1];

        // ---- 32x32 ze tile per wave: 4 accumulators, 8 k-steps of 32 ----
        v8f acc00 = {}, acc01 = {}, acc10 = {}, acc11 = {};
        #pragma unroll
        for (int ks = 0; ks < 8; ++ks) {
            union { v16h v; v8h h[2]; } A0, A1, B0, B1;
            const int k0 = ks * 32 + khalf;
            A0.h[0] = *(const v8h*)&zt[arow0][k0];
            A0.h[1] = *(const v8h*)&zt[arow0][k0 + 16];
            A1.h[0] = *(const v8h*)&zt[arow1][k0];
            A1.h[1] = *(const v8h*)&zt[arow1][k0 + 16];
            const int bcol = koff + (lane & 15);
            #pragma unroll
            for (int s = 0; s < 16; ++s) {
                const int bdim = ks * 32 + khalf + (s & 7) + ((s >> 3) << 4);
                B0.v[s] = cbc[bdim][bcol];
                B1.v[s] = cbc[bdim][bcol + 16];
            }
            acc00 = __builtin_amdgcn_wmma_f32_16x16x32_f16(
                false, A0.v, false, B0.v, (short)0, acc00, false, false);
            acc01 = __builtin_amdgcn_wmma_f32_16x16x32_f16(
                false, A0.v, false, B1.v, (short)0, acc01, false, false);
            acc10 = __builtin_amdgcn_wmma_f32_16x16x32_f16(
                false, A1.v, false, B0.v, (short)0, acc10, false, false);
            acc11 = __builtin_amdgcn_wmma_f32_16x16x32_f16(
                false, A1.v, false, B1.v, (short)0, acc11, false, false);
        }

        // ---- running argmin: score = ||e||^2 - 2 * z.e ----
        const int cg0 = it * KITER + cgLane;       // codes of B0 column
        const float e20 = e2g[cg0];
        const float e21 = e2g[cg0 + 16];           // codes of B1 column
        #pragma unroll
        for (int j = 0; j < 8; ++j) {
            {   // row tile a=0
                float s0 = __builtin_fmaf(-2.0f, acc00[j], e20);
                float s1 = __builtin_fmaf(-2.0f, acc01[j], e21);
                float s = s0; int ci = cg0;
                if (s1 < s0) { s = s1; ci = cg0 + 16; }
                if (s < runMin[j]) { runMin[j] = s; runIdx[j] = ci; }
            }
            {   // row tile a=1
                float s0 = __builtin_fmaf(-2.0f, acc10[j], e20);
                float s1 = __builtin_fmaf(-2.0f, acc11[j], e21);
                float s = s0; int ci = cg0;
                if (s1 < s0) { s = s1; ci = cg0 + 16; }
                if (s < runMin[8 + j]) { runMin[8 + j] = s; runIdx[8 + j] = ci; }
            }
        }
        // no trailing barrier: next iteration's wait+barrier provides it
    }

    // ---- reduce across the 16 N-lanes of each half ----
    #pragma unroll
    for (int t = 0; t < 16; ++t) {
        float m = runMin[t]; int mi = runIdx[t];
        #pragma unroll
        for (int off = 1; off < 16; off <<= 1) {
            float om = __shfl_xor(m, off);
            int   oi = __shfl_xor(mi, off);
            if (om < m || (om == m && oi < mi)) { m = om; mi = oi; }
        }
        runMin[t] = m; runIdx[t] = mi;
    }
    if ((lane & 15) == 0) {
        const int half = lane >> 4;   // 0: rows +j, 1: rows +8+j
        #pragma unroll
        for (int t = 0; t < 16; ++t) {
            const int a = t >> 3, j = t & 7;
            const int row = mgroup * 32 + a * 16 + half * 8 + j;
            redV[row][kslot] = runMin[t];
            redI[row][kslot] = runIdx[t];
        }
    }
    __syncthreads();

    // ---- reduce across the 2 code-slot waves, emit indices ----
    if (tid < ROWS) {
        float m = redV[tid][0]; int mi = redI[tid][0];
        float om = redV[tid][1]; int oi = redI[tid][1];
        if (om < m || (om == m && oi < mi)) { m = om; mi = oi; }
        fidx[tid] = mi;
        out[ZQ_ELEMS + 1 + p0 + tid] = (float)mi;   // idx output
    }
    __syncthreads();

    // ---- gather z_q (fp32), scatter to [B,D,H,W], accumulate loss ----
    float partial = 0.0f;
    {
        const int r  = tid & 127;
        const int dg = tid >> 7;
        const int p  = p0 + r;
        const int b  = p >> 12;
        const int q  = p & (HWSZ - 1);
        const int code = fidx[r];
        const float* crow = cb + (size_t)code * DIM;
        const float* zrow = z + (size_t)b * DIM * HWSZ + q;
        float* orow = out + (size_t)b * DIM * HWSZ + q;
        #pragma unroll 4
        for (int i = 0; i < 128; ++i) {
            int d = dg * 128 + i;
            float e  = crow[d];
            float zv = zrow[(size_t)d * HWSZ];
            orow[(size_t)d * HWSZ] = e;              // z_q_st == z_q numerically
            float df = e - zv;
            partial = __builtin_fmaf(df, df, partial);
        }
    }
    #pragma unroll
    for (int off = 16; off > 0; off >>= 1) partial += __shfl_down(partial, off);
    if (lane == 0) lred[wave] = partial;
    __syncthreads();
    if (tid == 0) {
        float s = 0.0f;
        #pragma unroll
        for (int w = 0; w < 8; ++w) s += lred[w];
        // vq_loss = mean + BETA*mean = 1.25 * sum/(N*D)
        atomicAdd(out + ZQ_ELEMS, s * (1.25f / ((float)NTOK * (float)DIM)));
    }
}

// ---------------------------------------------------------------------------
extern "C" void kernel_launch(void* const* d_in, const int* in_sizes, int n_in,
                              void* d_out, int out_size, void* d_ws, size_t ws_size,
                              hipStream_t stream) {
    const float* z  = (const float*)d_in[0];   // [8,256,64,64] fp32
    const float* cb = (const float*)d_in[1];   // [4096,256] fp32
    float* out = (float*)d_out;

    _Float16* cbhT = (_Float16*)d_ws;                                  // 2 MB, [dim][code]
    float*    e2   = (float*)((char*)d_ws + (size_t)NCODE * DIM * sizeof(_Float16));
    float*    loss_slot = out + ZQ_ELEMS;

    vq_prep<<<NCODE, 64, 0, stream>>>(cb, cbhT, e2, loss_slot);
    vq_main<<<NTOK / ROWS, 256, 0, stream>>>(z, cb, cbhT, e2, out);
}
